// Attention_79087527788690
// MI455X (gfx1250) — matrix operand order repaired and checked by reading, not compiled, not run on hardware
//
#include <hip/hip_runtime.h>
#include <hip/hip_bf16.h>
#include <math.h>

typedef __bf16 bf16_t;
typedef __attribute__((ext_vector_type(16))) __bf16 v16bf;
typedef __attribute__((ext_vector_type(8)))  __bf16 v8bf;
typedef __attribute__((ext_vector_type(8)))  float  v8f;

union FragU {
    v16bf v;
    struct { v8bf lo; v8bf hi; } p;
};

#define WMMA_BF16(a, b, c) \
    __builtin_amdgcn_wmma_f32_16x16x32_bf16(false, (a), false, (b), (short)0, (c), false, false)

// ---------------------------------------------------------------------------
// bf16 WMMA GEMM:  C[M][N] (+= bias) = A[M][K] @ (Bt[N][K])^T
// Both operands stored rows-of-K. Grid: (N/64, M/128), block = 128 (4 waves).
// Each wave: 32 rows x 64 cols (2 A frags x 4 B frags = 8 WMMAs / k-step).
// Software-pipelined: the steady-state loop body is branch-free — it loads
// the NEXT k-step's fragments, then issues 8 WMMAs against the current ones;
// the final k-step is peeled so no per-iteration branch/copies are needed.
// Requires: M mult of 128; N mult of 64; K mult of 32 (K >= 32).
// ---------------------------------------------------------------------------
__global__ __launch_bounds__(128)
void gemm_bf16_wmma(const bf16_t* __restrict__ A, int lda,
                    const bf16_t* __restrict__ Bt, int ldb,
                    float* __restrict__ Cf, bf16_t* __restrict__ Cb, int ldc,
                    int K, const float* __restrict__ bias) {
    const int lane = threadIdx.x & 31;
    const int wave = threadIdx.x >> 5;
    const int nloc = lane & 15;
    const int hi   = lane >> 4;          // half-wave select
    const int row_blk = blockIdx.y * 128 + wave * 32;
    const int col_blk = blockIdx.x * 64;

    // Per-lane base pointers (A rows for the two 16-row sets, B column rows)
    const bf16_t* arow0 = A + (size_t)(row_blk + nloc) * lda + hi * 8;
    const bf16_t* arow1 = arow0 + (size_t)16 * lda;
    const bf16_t* bbase = Bt + (size_t)(col_blk + nloc) * ldb + hi * 8;
    const size_t  bstep = (size_t)16 * ldb;

    auto ldA = [&](const bf16_t* p, int k) -> v16bf {
        FragU f;
        f.p.lo = *(const v8bf*)(p + k);
        f.p.hi = *(const v8bf*)(p + k + 16);
        return f.v;
    };
    auto ldB = [&](int t, int k) -> v16bf {
        const bf16_t* p = bbase + (size_t)t * bstep + k;
        FragU f;
        f.p.lo = *(const v8bf*)(p);
        f.p.hi = *(const v8bf*)(p + 16);
        return f.v;
    };

    v8f acc[8] = {};

    // Prologue: fragments for k = 0
    v16bf a0 = ldA(arow0, 0), a1 = ldA(arow1, 0);
    v16bf b0 = ldB(0, 0), b1 = ldB(1, 0), b2 = ldB(2, 0), b3 = ldB(3, 0);

    // Steady state: branch-free body; last k-step peeled below.
    for (int k = 32; k < K; k += 32) {
        const v16bf na0 = ldA(arow0, k);
        const v16bf na1 = ldA(arow1, k);
        const v16bf nb0 = ldB(0, k);
        const v16bf nb1 = ldB(1, k);
        const v16bf nb2 = ldB(2, k);
        const v16bf nb3 = ldB(3, k);
        acc[0] = WMMA_BF16(a0, b0, acc[0]);
        acc[1] = WMMA_BF16(a0, b1, acc[1]);
        acc[2] = WMMA_BF16(a0, b2, acc[2]);
        acc[3] = WMMA_BF16(a0, b3, acc[3]);
        acc[4] = WMMA_BF16(a1, b0, acc[4]);
        acc[5] = WMMA_BF16(a1, b1, acc[5]);
        acc[6] = WMMA_BF16(a1, b2, acc[6]);
        acc[7] = WMMA_BF16(a1, b3, acc[7]);
        a0 = na0; a1 = na1; b0 = nb0; b1 = nb1; b2 = nb2; b3 = nb3;
    }
    // Epilogue: final k-step
    acc[0] = WMMA_BF16(a0, b0, acc[0]);
    acc[1] = WMMA_BF16(a0, b1, acc[1]);
    acc[2] = WMMA_BF16(a0, b2, acc[2]);
    acc[3] = WMMA_BF16(a0, b3, acc[3]);
    acc[4] = WMMA_BF16(a1, b0, acc[4]);
    acc[5] = WMMA_BF16(a1, b1, acc[5]);
    acc[6] = WMMA_BF16(a1, b2, acc[6]);
    acc[7] = WMMA_BF16(a1, b3, acc[7]);

#pragma unroll
    for (int s = 0; s < 2; ++s) {
#pragma unroll
        for (int t = 0; t < 4; ++t) {
            const int col = col_blk + t * 16 + nloc;
            const float bv = bias ? bias[col] : 0.0f;
            const v8f av = acc[s * 4 + t];
#pragma unroll
            for (int r = 0; r < 8; ++r) {
                const int row = row_blk + s * 16 + r + 8 * hi;
                const float v = av[r] + bv;
                if (Cb) Cb[(size_t)row * ldc + col] = (bf16_t)v;
                else    Cf[(size_t)row * ldc + col] = v;
            }
        }
    }
}

// ---------------------------------------------------------------------------
// fp32 -> bf16 elementwise convert
// ---------------------------------------------------------------------------
__global__ void convert_f32_bf16(const float* __restrict__ in,
                                 bf16_t* __restrict__ out, int n) {
    int i = blockIdx.x * blockDim.x + threadIdx.x;
    if (i < n) out[i] = (bf16_t)in[i];
}

// ---------------------------------------------------------------------------
// fp32 [R][C] -> bf16 [C][R] tiled transpose
// ---------------------------------------------------------------------------
__global__ void transpose_f32_bf16(const float* __restrict__ in,
                                   bf16_t* __restrict__ out, int R, int C) {
    __shared__ float tile[32][33];
    const int c0 = blockIdx.x * 32;
    const int r0 = blockIdx.y * 32;
    const int tx = threadIdx.x, ty = threadIdx.y;  // (32, 8)
#pragma unroll
    for (int s = 0; s < 32; s += 8) {
        int r = r0 + ty + s, c = c0 + tx;
        tile[ty + s][tx] = (r < R && c < C) ? in[(size_t)r * C + c] : 0.0f;
    }
    __syncthreads();
#pragma unroll
    for (int s = 0; s < 32; s += 8) {
        int r = r0 + tx, c = c0 + ty + s;
        if (c < C && r < R) out[(size_t)c * R + r] = (bf16_t)tile[tx][ty + s];
    }
}

// ---------------------------------------------------------------------------
// Positional embedding P[4096][192] (row 4095 = zero pad), bf16.
// One wave (32 lanes) per position; lane i handles feature triple i (nb=32).
// ---------------------------------------------------------------------------
__global__ __launch_bounds__(32)
void pos_embed_kernel(bf16_t* __restrict__ P) {
    const int pos = blockIdx.x;
    const int i = threadIdx.x;
    const size_t base = (size_t)pos * 192;
    if (pos >= 4095) {
        for (int t = i; t < 192; t += 32) P[base + t] = (bf16_t)0.0f;
        return;
    }
    const float d = (float)(pos - 2047);
    const float absd = fabsf(d);

    // exponential features: half_life = 2^linspace(3, 11, 32)
    const float hl = exp2f(3.0f + 8.0f * (float)i / 31.0f);
    const float fe = expf(-0.6931471805599453f / hl * absd);

    // central-mask features
    const float cw = exp2f((float)(i + 1)) - 1.0f;
    const float fc = (cw > absd) ? 1.0f : 0.0f;

    // gamma features: mean = linspace(64, 2048, 32), stddev = 32
    const float mean = 64.0f + (2048.0f - 64.0f) * (float)i / 31.0f;
    const float conc = (mean / 32.0f) * (mean / 32.0f);
    const float rate = mean / 1024.0f;
    const float lu = (absd > 0.0f) ? ((conc - 1.0f) * logf(absd) - rate * absd)
                                   : -INFINITY;  // xlogy with conc-1 > 0
    const float ln_ = lgammaf(conc) - conc * logf(rate);
    float pr = expf(lu - ln_) + 1e-8f;
    float mx = pr;
#pragma unroll
    for (int off = 16; off > 0; off >>= 1)
        mx = fmaxf(mx, __shfl_xor(mx, off, 32));
    const float fg = pr / mx;

    const float sgn = (d > 0.0f) ? 1.0f : ((d < 0.0f) ? -1.0f : 0.0f);
    P[base +       i] = (bf16_t)fe;
    P[base +  32 + i] = (bf16_t)fc;
    P[base +  64 + i] = (bf16_t)fg;
    P[base +  96 + i] = (bf16_t)(sgn * fe);
    P[base + 128 + i] = (bf16_t)(sgn * fc);
    P[base + 160 + i] = (bf16_t)(sgn * fg);
}

// ---------------------------------------------------------------------------
// qc = bf16(q*scale + rel_content_bias), qr = bf16(q*scale + rel_pos_bias)
// ---------------------------------------------------------------------------
__global__ void qbias_kernel(const float* __restrict__ qf,
                             const float* __restrict__ rcb,
                             const float* __restrict__ rpb,
                             bf16_t* __restrict__ qc,
                             bf16_t* __restrict__ qr, int n) {
    int idx = blockIdx.x * blockDim.x + threadIdx.x;
    if (idx >= n) return;
    const int col = idx & 511;           // 512 columns = H*DK
    const float qv = qf[idx] * 0.125f;   // 64^-0.5
    qc[idx] = (bf16_t)(qv + rcb[col]);
    qr[idx] = (bf16_t)(qv + rpb[col]);
}

// ---------------------------------------------------------------------------
// Softmax over content[i][j] + rel[i][2047 + j - i]  (fused relative_shift),
// writes bf16 attention. One block of 256 threads per row.
// ---------------------------------------------------------------------------
__global__ __launch_bounds__(256)
void softmax_rel_kernel(const float* __restrict__ content,
                        const float* __restrict__ relfull,
                        bf16_t* __restrict__ attn) {
    __shared__ float vals[2048];
    __shared__ float red[256];
    const int i = blockIdx.x;
    const int tid = threadIdx.x;
    const float* crow = content + (size_t)i * 2048;
    const float* rrow = relfull + (size_t)i * 4096 + (2047 - i);

    float lmax = -3.4e38f;
    for (int j = tid; j < 2048; j += 256) {
        float v = crow[j] + rrow[j];
        vals[j] = v;
        lmax = fmaxf(lmax, v);
    }
    red[tid] = lmax;
    __syncthreads();
    for (int s = 128; s > 0; s >>= 1) {
        if (tid < s) red[tid] = fmaxf(red[tid], red[tid + s]);
        __syncthreads();
    }
    const float mx = red[0];
    __syncthreads();

    float lsum = 0.0f;
    for (int j = tid; j < 2048; j += 256) {
        float e = __expf(vals[j] - mx);
        vals[j] = e;
        lsum += e;
    }
    red[tid] = lsum;
    __syncthreads();
    for (int s = 128; s > 0; s >>= 1) {
        if (tid < s) red[tid] += red[tid + s];
        __syncthreads();
    }
    const float inv = 1.0f / red[0];
    bf16_t* arow = attn + (size_t)i * 2048;
    for (int j = tid; j < 2048; j += 256) arow[j] = (bf16_t)(vals[j] * inv);
}

// ---------------------------------------------------------------------------
extern "C" void kernel_launch(void* const* d_in, const int* in_sizes, int n_in,
                              void* d_out, int out_size, void* d_ws, size_t ws_size,
                              hipStream_t stream) {
    const float* x    = (const float*)d_in[0];
    const float* Wq   = (const float*)d_in[1];
    const float* Wk   = (const float*)d_in[2];
    const float* Wv   = (const float*)d_in[3];
    const float* Wrel = (const float*)d_in[4];
    const float* rcb  = (const float*)d_in[5];
    const float* rpb  = (const float*)d_in[6];
    const float* Wo   = (const float*)d_in[7];
    const float* bo   = (const float*)d_in[8];
    float* out = (float*)d_out;

    const int N = 2048, D = 1536, H = 8, DK = 64, DV = 192;
    const int HDK = 512, HDV = 1536, LP = 4096, NRF = 192;

    char* base = (char*)d_ws;
    size_t off = 0;
    auto alloc = [&](size_t bytes) -> void* {
        off = (off + 255) & ~(size_t)255;
        void* p = base + off;
        off += bytes;
        return p;
    };
    bf16_t* xb   = (bf16_t*)alloc((size_t)N * D * 2);
    bf16_t* WqT  = (bf16_t*)alloc((size_t)HDK * D * 2);
    bf16_t* WkT  = (bf16_t*)alloc((size_t)HDK * D * 2);
    bf16_t* WvT  = (bf16_t*)alloc((size_t)HDV * D * 2);
    bf16_t* WrT  = (bf16_t*)alloc((size_t)HDK * NRF * 2);
    bf16_t* WoT  = (bf16_t*)alloc((size_t)D * HDV * 2);
    bf16_t* Pb   = (bf16_t*)alloc((size_t)LP * NRF * 2);
    float*  qf   = (float*) alloc((size_t)N * HDK * 4);
    bf16_t* qc   = (bf16_t*)alloc((size_t)N * HDK * 2);
    bf16_t* qr   = (bf16_t*)alloc((size_t)N * HDK * 2);
    bf16_t* kb   = (bf16_t*)alloc((size_t)N * HDK * 2);
    float*  vf   = (float*) alloc((size_t)N * HDV * 4);
    bf16_t* vT   = (bf16_t*)alloc((size_t)HDV * N * 2);
    bf16_t* relk = (bf16_t*)alloc((size_t)LP * HDK * 2);
    float*  cont = (float*) alloc((size_t)N * N * 4);
    float*  relf = (float*) alloc((size_t)N * LP * 4);
    bf16_t* attn = (bf16_t*)alloc((size_t)N * N * 2);
    float*  aof  = (float*) alloc((size_t)N * HDV * 4);
    bf16_t* aob  = (bf16_t*)alloc((size_t)N * HDV * 2);

    const dim3 tb(32, 8);

    // operand preparation (fp32 -> bf16, weights transposed to [N][K])
    convert_f32_bf16<<<(N * D + 255) / 256, 256, 0, stream>>>(x, xb, N * D);
    transpose_f32_bf16<<<dim3(HDK / 32, D / 32), tb, 0, stream>>>(Wq, WqT, D, HDK);
    transpose_f32_bf16<<<dim3(HDK / 32, D / 32), tb, 0, stream>>>(Wk, WkT, D, HDK);
    transpose_f32_bf16<<<dim3(HDV / 32, D / 32), tb, 0, stream>>>(Wv, WvT, D, HDV);
    transpose_f32_bf16<<<dim3(HDK / 32, NRF / 32), tb, 0, stream>>>(Wrel, WrT, NRF, HDK);
    transpose_f32_bf16<<<dim3(D / 32, HDV / 32), tb, 0, stream>>>(Wo, WoT, HDV, D);
    pos_embed_kernel<<<LP, 32, 0, stream>>>(Pb);

    // projections (WMMA)
    gemm_bf16_wmma<<<dim3(HDK / 64, N / 128), 128, 0, stream>>>(
        xb, D, WqT, D, qf, nullptr, HDK, D, nullptr);
    gemm_bf16_wmma<<<dim3(HDK / 64, N / 128), 128, 0, stream>>>(
        xb, D, WkT, D, nullptr, kb, HDK, D, nullptr);
    gemm_bf16_wmma<<<dim3(HDV / 64, N / 128), 128, 0, stream>>>(
        xb, D, WvT, D, vf, nullptr, HDV, D, nullptr);
    gemm_bf16_wmma<<<dim3(HDK / 64, LP / 128), 128, 0, stream>>>(
        Pb, NRF, WrT, NRF, nullptr, relk, HDK, NRF, nullptr);

    qbias_kernel<<<(N * HDK + 255) / 256, 256, 0, stream>>>(qf, rcb, rpb, qc, qr, N * HDK);
    transpose_f32_bf16<<<dim3(HDV / 32, N / 32), tb, 0, stream>>>(vf, vT, N, HDV);

    // per-head attention (reuses cont/relf/attn scratch; ~56MB live)
    for (int h = 0; h < H; ++h) {
        gemm_bf16_wmma<<<dim3(N / 64, N / 128), 128, 0, stream>>>(
            qc + h * DK, HDK, kb + h * DK, HDK, cont, nullptr, N, DK, nullptr);
        gemm_bf16_wmma<<<dim3(LP / 64, N / 128), 128, 0, stream>>>(
            qr + h * DK, HDK, relk + h * DK, HDK, relf, nullptr, LP, DK, nullptr);
        softmax_rel_kernel<<<N, 256, 0, stream>>>(cont, relf, attn);
        gemm_bf16_wmma<<<dim3(DV / 64, N / 128), 128, 0, stream>>>(
            attn, N, vT + (size_t)h * DV * N, N, aof + h * DV, nullptr, HDV, N, nullptr);
    }

    // output projection + bias -> d_out (fp32)
    convert_f32_bf16<<<(N * HDV + 255) / 256, 256, 0, stream>>>(aof, aob, N * HDV);
    gemm_bf16_wmma<<<dim3(HDV / 64, N / 128), 128, 0, stream>>>(
        aob, HDV, WoT, HDV, out, nullptr, HDV, HDV, bo);
}
